// AttentionLayer_80032420594410
// MI455X (gfx1250) — matrix operand order repaired and checked by reading, not compiled, not run on hardware
//
#include <hip/hip_runtime.h>

// ---- vector / fragment types -------------------------------------------------
typedef __bf16 v16bf __attribute__((ext_vector_type(16)));
typedef float  v8f   __attribute__((ext_vector_type(8)));
typedef float  f32x4 __attribute__((ext_vector_type(4)));
typedef unsigned short u16x8 __attribute__((ext_vector_type(8)));
typedef unsigned short u16x4 __attribute__((ext_vector_type(4)));

union Frag16 { u16x8 u[2]; v16bf v; };        // 32B: A or B fragment (16 bf16)
union V8     { v8f v; f32x4 q[2]; float f[8]; };

__device__ __forceinline__ unsigned short f32_to_bf16(float x) {
  unsigned int u = __float_as_uint(x);
  u += 0x7fffu + ((u >> 16) & 1u);            // round-to-nearest-even
  return (unsigned short)(u >> 16);
}

// WGP-scope prefetch (scope field omitted -> SCOPE_CU/WGP = 0, which fills all
// cache levels).  __builtin_prefetch can only reach SE/SYS scopes, which the
// ISA defines as GL2-only prefetch -- useless here since data is L2-resident.
__device__ __forceinline__ void prefetch_wgp(const void* p) {
  asm volatile("global_prefetch_b8 %0, off" :: "v"(p));
}

// Problem dims (fixed by the reference)
#define BATCH 8
#define SEQ   2048
#define DIM   512
#define BR    16      // query rows per workgroup
#define CK    32      // keys per chunk (WMMA K for bf16)
#define NW    4       // waves per workgroup
#define DW    128     // head-dim slice per wave

// =============================================================================
// Kernel 0: fp32 -> bf16, producing row-major Xrow[b][s][d] AND transposed
// Xtr[b][d][s].  32x32 tile transpose through LDS; both outputs coalesced.
// =============================================================================
__launch_bounds__(256, 4)
__global__ void cvt_bf16_dual(const float* __restrict__ X,
                              unsigned short* __restrict__ Xrow,
                              unsigned short* __restrict__ Xtr) {
  __shared__ unsigned short T[32][33];        // padded to dodge bank conflicts
  const int b  = blockIdx.z;
  const int s0 = blockIdx.y * 32;
  const int d0 = blockIdx.x * 32;
  const int t  = threadIdx.x;
  const int r  = t >> 3;                      // 0..31
  const int c4 = (t & 7) * 4;                 // 0,4,...,28

  {
    f32x4 f = *(const f32x4*)(X + ((size_t)(b * SEQ + s0 + r) * DIM) + d0 + c4);
    u16x4 h;
    h[0] = f32_to_bf16(f[0]); h[1] = f32_to_bf16(f[1]);
    h[2] = f32_to_bf16(f[2]); h[3] = f32_to_bf16(f[3]);
    *(u16x4*)(Xrow + ((size_t)(b * SEQ + s0 + r) * DIM) + d0 + c4) = h;
    T[r][c4 + 0] = h[0]; T[r][c4 + 1] = h[1];
    T[r][c4 + 2] = h[2]; T[r][c4 + 3] = h[3];
  }
  __syncthreads();
  {
    u16x4 h;
    h[0] = T[c4 + 0][r]; h[1] = T[c4 + 1][r];
    h[2] = T[c4 + 2][r]; h[3] = T[c4 + 3][r];
    *(u16x4*)(Xtr + ((size_t)(b * DIM + d0 + r) * SEQ) + s0 + c4) = h;
  }
}

// =============================================================================
// Kernel 1 (fast path): FA2-style attention.  All K/V fragments are direct
// global b128 loads from the L2-resident bf16 copies (192MB L2 >> 32MB data).
// LDS holds only the Q tile, the cross-wave S reduction, and the P tile.
// =============================================================================
__launch_bounds__(128, 2)
__global__ void attn_fa2_direct(const unsigned short* __restrict__ Xrow,
                                const unsigned short* __restrict__ Xtr,
                                float* __restrict__ O) {
  __shared__ alignas(16) unsigned short Qs[BR][DIM];        // 16 KB
  __shared__ alignas(16) float          Sred[NW][2][32][8]; //  8 KB
  __shared__ alignas(16) unsigned short Ps[BR][CK];         //  1 KB

  const int b    = blockIdx.y;
  const int m0   = blockIdx.x * BR;
  const int tid  = threadIdx.x;
  const int wave = tid >> 5;
  const int lane = tid & 31;
  const int half = lane >> 4;
  const int l16  = lane & 15;

  const unsigned short* Xr = Xrow + (size_t)b * SEQ * DIM;
  const unsigned short* Xt = Xtr + (size_t)b * DIM * SEQ;
  float* Ob = O + (size_t)b * SEQ * DIM;

  // stage Q tile (bf16 copy, vector DS stores)
  for (int i = tid; i < BR * (DIM / 8); i += 128) {
    const int row = i / (DIM / 8);
    const int g   = (i % (DIM / 8)) * 8;
    *(u16x8*)&Qs[row][g] = *(const u16x8*)(Xr + (size_t)(m0 + row) * DIM + g);
  }

  v8f acc[8];
  #pragma unroll
  for (int t = 0; t < 8; ++t) acc[t] = (v8f){0,0,0,0,0,0,0,0};
  float m_run[8], l_run[8];
  #pragma unroll
  for (int r = 0; r < 8; ++r) { m_run[r] = -3.0e38f; l_run[r] = 0.0f; }

  __syncthreads();

  const int dbase = wave * DW;
  const int kb_h  = half ? 16 : 0;

  for (int j0 = 0; j0 < SEQ; j0 += CK) {
    // WGP-scope prefetch of next chunk's operands into the near caches.
    if (j0 + CK < SEQ) {
      prefetch_wgp(Xr + (size_t)(j0 + CK + lane) * DIM + dbase);
      prefetch_wgp(Xr + (size_t)(j0 + CK + lane) * DIM + dbase + 64);
      prefetch_wgp(Xt + (size_t)(dbase + lane * 4) * SEQ + j0 + CK);
    }

    // ---- partial S = Q * K^T over this wave's 128-dim slice -----------------
    v8f s0 = (v8f){0,0,0,0,0,0,0,0};
    v8f s1 = (v8f){0,0,0,0,0,0,0,0};
    const unsigned short* Kr0 = Xr + (size_t)(j0 + l16) * DIM;        // keys 0-15
    const unsigned short* Kr1 = Kr0 + (size_t)16 * DIM;               // keys 16-31
    #pragma unroll
    for (int c = 0; c < 4; ++c) {
      const int d0 = dbase + c * 32;
      Frag16 a;                                   // A: Q rows, lane = row m
      const int ab = d0 + (half ? 16 : 0);
      a.u[0] = *(const u16x8*)&Qs[l16][ab];
      a.u[1] = *(const u16x8*)&Qs[l16][ab + 8];
      const int kb = d0 + kb_h;                   // B: K^T, lane = key col
      Frag16 b0, b1;
      b0.u[0] = *(const u16x8*)(Kr0 + kb);
      b0.u[1] = *(const u16x8*)(Kr0 + kb + 8);
      b1.u[0] = *(const u16x8*)(Kr1 + kb);
      b1.u[1] = *(const u16x8*)(Kr1 + kb + 8);
      s0 = __builtin_amdgcn_wmma_f32_16x16x32_bf16(false, a.v, false, b0.v,
                                                   (short)0, s0, false, false);
      s1 = __builtin_amdgcn_wmma_f32_16x16x32_bf16(false, a.v, false, b1.v,
                                                   (short)0, s1, false, false);
    }

    // ---- cross-wave reduction of partial S ----------------------------------
    {
      V8 w0, w1; w0.v = s0; w1.v = s1;
      *(f32x4*)&Sred[wave][0][lane][0] = w0.q[0];
      *(f32x4*)&Sred[wave][0][lane][4] = w0.q[1];
      *(f32x4*)&Sred[wave][1][lane][0] = w1.q[0];
      *(f32x4*)&Sred[wave][1][lane][4] = w1.q[1];
    }
    __syncthreads();
    V8 fs0, fs1;
    fs0.v = (v8f){0,0,0,0,0,0,0,0};
    fs1.v = (v8f){0,0,0,0,0,0,0,0};
    #pragma unroll
    for (int w = 0; w < NW; ++w) {
      fs0.q[0] += *(const f32x4*)&Sred[w][0][lane][0];
      fs0.q[1] += *(const f32x4*)&Sred[w][0][lane][4];
      fs1.q[0] += *(const f32x4*)&Sred[w][1][lane][0];
      fs1.q[1] += *(const f32x4*)&Sred[w][1][lane][4];
    }

    // ---- online softmax (rows live in 16-lane halves) -----------------------
    float p0[8], p1[8], scale[8];
    #pragma unroll
    for (int r = 0; r < 8; ++r) {
      float mx = fmaxf(fs0.f[r], fs1.f[r]);
      mx = fmaxf(mx, __shfl_xor(mx, 1, 32));
      mx = fmaxf(mx, __shfl_xor(mx, 2, 32));
      mx = fmaxf(mx, __shfl_xor(mx, 4, 32));
      mx = fmaxf(mx, __shfl_xor(mx, 8, 32));
      const float mn = fmaxf(m_run[r], mx);
      scale[r] = __expf(m_run[r] - mn);
      p0[r] = __expf(fs0.f[r] - mn);
      p1[r] = __expf(fs1.f[r] - mn);
      float rs = p0[r] + p1[r];
      rs += __shfl_xor(rs, 1, 32);
      rs += __shfl_xor(rs, 2, 32);
      rs += __shfl_xor(rs, 4, 32);
      rs += __shfl_xor(rs, 8, 32);
      l_run[r] = l_run[r] * scale[r] + rs;
      m_run[r] = mn;
    }
    #pragma unroll
    for (int t = 0; t < 8; ++t) {
      V8 a; a.v = acc[t];
      #pragma unroll
      for (int r = 0; r < 8; ++r) a.f[r] *= scale[r];
      acc[t] = a.v;
    }

    // ---- publish P tile as bf16, A-fragment row-major -----------------------
    if (wave == 0) {
      #pragma unroll
      for (int r = 0; r < 8; ++r) {
        const int row = r + half * 8;
        Ps[row][l16]      = f32_to_bf16(p0[r]);
        Ps[row][16 + l16] = f32_to_bf16(p1[r]);
      }
    }
    __syncthreads();

    // ---- O slice += P * V: B-fragments straight from transposed global ------
    Frag16 pa;
    {
      const int pb = half ? 16 : 0;
      pa.u[0] = *(const u16x8*)&Ps[l16][pb];
      pa.u[1] = *(const u16x8*)&Ps[l16][pb + 8];
    }
    const int krow = j0 + (half ? 16 : 0);
    #pragma unroll
    for (int t = 0; t < 8; ++t) {
      const int dcol = dbase + t * 16 + l16;
      const unsigned short* Vp = Xt + (size_t)dcol * SEQ + krow;
      Frag16 vb;
      vb.u[0] = *(const u16x8*)(Vp);
      vb.u[1] = *(const u16x8*)(Vp + 8);
      acc[t] = __builtin_amdgcn_wmma_f32_16x16x32_bf16(false, pa.v, false, vb.v,
                                                       (short)0, acc[t], false, false);
    }
  }

  // ---- epilogue: normalize and store ----------------------------------------
  float inv[8];
  #pragma unroll
  for (int r = 0; r < 8; ++r) inv[r] = 1.0f / l_run[r];
  #pragma unroll
  for (int t = 0; t < 8; ++t) {
    V8 a; a.v = acc[t];
    #pragma unroll
    for (int r = 0; r < 8; ++r) {
      const int row = r + half * 8;
      Ob[(size_t)(m0 + row) * DIM + dbase + t * 16 + l16] = a.f[r] * inv[r];
    }
  }
}

// =============================================================================
// Kernel 2 (fallback, used only if d_ws is too small): inline fp32->bf16
// conversion with LDS staging.
// =============================================================================
__launch_bounds__(128, 1)
__global__ void attn_fa2_fallback(const float* __restrict__ X,
                                  float* __restrict__ O) {
  __shared__ alignas(16) unsigned short Qs[BR][DIM];
  __shared__ alignas(16) unsigned short Vs[CK][DIM];
  __shared__ alignas(16) unsigned short Vt[DIM][CK];
  __shared__ alignas(16) float          Sred[NW][2][32][8];
  __shared__ alignas(16) unsigned short Ps[BR][CK];

  const int b    = blockIdx.y;
  const int m0   = blockIdx.x * BR;
  const int tid  = threadIdx.x;
  const int wave = tid >> 5;
  const int lane = tid & 31;
  const int half = lane >> 4;
  const int l16  = lane & 15;

  const float* Xb = X + (size_t)b * SEQ * DIM;
  float*       Ob = O + (size_t)b * SEQ * DIM;

  for (int i = tid; i < BR * (DIM / 4); i += 128) {
    const int row = i / (DIM / 4);
    const int g   = (i % (DIM / 4)) * 4;
    f32x4 f = *(const f32x4*)(Xb + (size_t)(m0 + row) * DIM + g);
    u16x4 h;
    h[0] = f32_to_bf16(f[0]); h[1] = f32_to_bf16(f[1]);
    h[2] = f32_to_bf16(f[2]); h[3] = f32_to_bf16(f[3]);
    *(u16x4*)&Qs[row][g] = h;
  }

  v8f acc[8];
  #pragma unroll
  for (int t = 0; t < 8; ++t) acc[t] = (v8f){0,0,0,0,0,0,0,0};
  float m_run[8], l_run[8];
  #pragma unroll
  for (int r = 0; r < 8; ++r) { m_run[r] = -3.0e38f; l_run[r] = 0.0f; }

  for (int j0 = 0; j0 < SEQ; j0 += CK) {
    __syncthreads();
    for (int i = tid; i < CK * (DIM / 4); i += 128) {
      const int key = i / (DIM / 4);
      const int g   = (i % (DIM / 4)) * 4;
      f32x4 f = *(const f32x4*)(Xb + (size_t)(j0 + key) * DIM + g);
      u16x4 h;
      h[0] = f32_to_bf16(f[0]); h[1] = f32_to_bf16(f[1]);
      h[2] = f32_to_bf16(f[2]); h[3] = f32_to_bf16(f[3]);
      *(u16x4*)&Vs[key][g] = h;
      Vt[g + 0][key] = h[0]; Vt[g + 1][key] = h[1];
      Vt[g + 2][key] = h[2]; Vt[g + 3][key] = h[3];
    }
    __syncthreads();

    v8f s0 = (v8f){0,0,0,0,0,0,0,0};
    v8f s1 = (v8f){0,0,0,0,0,0,0,0};
    const int dbase = wave * DW;
    #pragma unroll
    for (int c = 0; c < 4; ++c) {
      const int d0 = dbase + c * 32;
      Frag16 a;
      const int ab = d0 + (half ? 16 : 0);
      a.u[0] = *(const u16x8*)&Qs[l16][ab];
      a.u[1] = *(const u16x8*)&Qs[l16][ab + 8];
      const int kb = d0 + (half ? 16 : 0);
      Frag16 b0, b1;
      b0.u[0] = *(const u16x8*)&Vs[l16][kb];
      b0.u[1] = *(const u16x8*)&Vs[l16][kb + 8];
      b1.u[0] = *(const u16x8*)&Vs[16 + l16][kb];
      b1.u[1] = *(const u16x8*)&Vs[16 + l16][kb + 8];
      s0 = __builtin_amdgcn_wmma_f32_16x16x32_bf16(false, a.v, false, b0.v,
                                                   (short)0, s0, false, false);
      s1 = __builtin_amdgcn_wmma_f32_16x16x32_bf16(false, a.v, false, b1.v,
                                                   (short)0, s1, false, false);
    }

    {
      V8 w0, w1; w0.v = s0; w1.v = s1;
      *(f32x4*)&Sred[wave][0][lane][0] = w0.q[0];
      *(f32x4*)&Sred[wave][0][lane][4] = w0.q[1];
      *(f32x4*)&Sred[wave][1][lane][0] = w1.q[0];
      *(f32x4*)&Sred[wave][1][lane][4] = w1.q[1];
    }
    __syncthreads();
    V8 fs0, fs1;
    fs0.v = (v8f){0,0,0,0,0,0,0,0};
    fs1.v = (v8f){0,0,0,0,0,0,0,0};
    #pragma unroll
    for (int w = 0; w < NW; ++w) {
      fs0.q[0] += *(const f32x4*)&Sred[w][0][lane][0];
      fs0.q[1] += *(const f32x4*)&Sred[w][0][lane][4];
      fs1.q[0] += *(const f32x4*)&Sred[w][1][lane][0];
      fs1.q[1] += *(const f32x4*)&Sred[w][1][lane][4];
    }

    float p0[8], p1[8], scale[8];
    #pragma unroll
    for (int r = 0; r < 8; ++r) {
      float mx = fmaxf(fs0.f[r], fs1.f[r]);
      mx = fmaxf(mx, __shfl_xor(mx, 1, 32));
      mx = fmaxf(mx, __shfl_xor(mx, 2, 32));
      mx = fmaxf(mx, __shfl_xor(mx, 4, 32));
      mx = fmaxf(mx, __shfl_xor(mx, 8, 32));
      const float mn = fmaxf(m_run[r], mx);
      scale[r] = __expf(m_run[r] - mn);
      p0[r] = __expf(fs0.f[r] - mn);
      p1[r] = __expf(fs1.f[r] - mn);
      float rs = p0[r] + p1[r];
      rs += __shfl_xor(rs, 1, 32);
      rs += __shfl_xor(rs, 2, 32);
      rs += __shfl_xor(rs, 4, 32);
      rs += __shfl_xor(rs, 8, 32);
      l_run[r] = l_run[r] * scale[r] + rs;
      m_run[r] = mn;
    }
    #pragma unroll
    for (int t = 0; t < 8; ++t) {
      V8 a; a.v = acc[t];
      #pragma unroll
      for (int r = 0; r < 8; ++r) a.f[r] *= scale[r];
      acc[t] = a.v;
    }

    if (wave == 0) {
      #pragma unroll
      for (int r = 0; r < 8; ++r) {
        const int row = r + half * 8;
        Ps[row][l16]      = f32_to_bf16(p0[r]);
        Ps[row][16 + l16] = f32_to_bf16(p1[r]);
      }
    }
    __syncthreads();

    Frag16 pa;
    {
      const int pb = half ? 16 : 0;
      pa.u[0] = *(const u16x8*)&Ps[l16][pb];
      pa.u[1] = *(const u16x8*)&Ps[l16][pb + 8];
    }
    const int krow = half ? 16 : 0;
    #pragma unroll
    for (int t = 0; t < 8; ++t) {
      const int dcol = wave * DW + t * 16 + l16;
      Frag16 vb;
      vb.u[0] = *(const u16x8*)&Vt[dcol][krow];
      vb.u[1] = *(const u16x8*)&Vt[dcol][krow + 8];
      acc[t] = __builtin_amdgcn_wmma_f32_16x16x32_bf16(false, pa.v, false, vb.v,
                                                       (short)0, acc[t], false, false);
    }
  }

  float inv[8];
  #pragma unroll
  for (int r = 0; r < 8; ++r) inv[r] = 1.0f / l_run[r];
  #pragma unroll
  for (int t = 0; t < 8; ++t) {
    V8 a; a.v = acc[t];
    #pragma unroll
    for (int r = 0; r < 8; ++r) {
      const int row = r + half * 8;
      Ob[(size_t)(m0 + row) * DIM + wave * DW + t * 16 + l16] = a.f[r] * inv[r];
    }
  }
}

// =============================================================================
extern "C" void kernel_launch(void* const* d_in, const int* in_sizes, int n_in,
                              void* d_out, int out_size, void* d_ws, size_t ws_size,
                              hipStream_t stream) {
  (void)in_sizes; (void)n_in; (void)out_size;
  const float* X = (const float*)d_in[0];
  float*       O = (float*)d_out;

  const size_t nelem = (size_t)BATCH * SEQ * DIM;
  const size_t need  = 2 * nelem * sizeof(unsigned short);   // 32 MB

  if (ws_size >= need) {
    unsigned short* Xrow = (unsigned short*)d_ws;
    unsigned short* Xtr  = Xrow + nelem;
    dim3 cgrid(DIM / 32, SEQ / 32, BATCH);                   // 16 x 64 x 8
    cvt_bf16_dual<<<cgrid, 256, 0, stream>>>(X, Xrow, Xtr);
    dim3 grid(SEQ / BR, BATCH);                              // 128 x 8
    attn_fa2_direct<<<grid, 128, 0, stream>>>(Xrow, Xtr, O);
  } else {
    dim3 grid(SEQ / BR, BATCH);
    attn_fa2_fallback<<<grid, 128, 0, stream>>>(X, O);
  }
}
